// TMLMambaLayer_50259707297889
// MI455X (gfx1250) — compile-verified, matrix operand order checked by reading, not run on hardware
//
#include <hip/hip_runtime.h>

typedef unsigned short u16;
typedef unsigned int   u32;
typedef __attribute__((ext_vector_type(16))) __bf16 v16bf;
typedef __attribute__((ext_vector_type(8)))  float  v8f;
typedef __attribute__((ext_vector_type(4)))  float  v4f;
typedef __attribute__((ext_vector_type(4)))  u32    v4u;

#define SEQ     1024
#define DM      512
#define DFF     2048
#define NBATCH  64
#define PAD     8
#define SEQP    (SEQ + 2 * PAD)   // padded rows for branch-free shifted GEMMs

union frag_u { v16bf v; v4u q[2]; };

__device__ __forceinline__ u16 f2bf(float f) {
    union { float f; u32 u; } cv; cv.f = f;
    u32 u = cv.u;
    u32 r = u + 0x7FFFu + ((u >> 16) & 1u);
    return (u16)(r >> 16);
}

// Lane fragment load (16x32 bf16 tile slice): two 16B loads per lane.
__device__ __forceinline__ v16bf load_frag(const u16* __restrict__ base, int row, int ld,
                                           int kbase, int half) {
    frag_u f;
    const u16* p = base + (size_t)row * ld + kbase + half * 8;
    f.q[0] = *(const v4u*)(p);
    f.q[1] = *(const v4u*)(p + 16);
    return f.v;
}

__device__ __forceinline__ v8f wmma_bf16(v16bf a, v16bf b, v8f c) {
    return __builtin_amdgcn_wmma_f32_16x16x32_bf16(false, a, false, b, (short)0, c,
                                                   false, false);
}

// One pipeline stage: A fragment + 4 B fragments (16x64 strip worth of operands)
struct Stage { v16bf a, b0, b1, b2, b3; };

__device__ __forceinline__ void load_stage(Stage& s,
                                           const u16* __restrict__ A, int arow, int lda,
                                           const u16* __restrict__ B, int bn, int ldb,
                                           int kb, int half) {
    s.a  = load_frag(A, arow, lda, kb, half);
    s.b0 = load_frag(B, bn,      ldb, kb, half);
    s.b1 = load_frag(B, bn + 16, ldb, kb, half);
    s.b2 = load_frag(B, bn + 32, ldb, kb, half);
    s.b3 = load_frag(B, bn + 48, ldb, kb, half);
}

__device__ __forceinline__ void fire(v8f acc[4], const Stage& s) {
    acc[0] = wmma_bf16(s.a, s.b0, acc[0]);
    acc[1] = wmma_bf16(s.a, s.b1, acc[1]);
    acc[2] = wmma_bf16(s.a, s.b2, acc[2]);
    acc[3] = wmma_bf16(s.a, s.b3, acc[3]);
}

// ---------------------------------------------------------------------------
// K0: alpha = softmax(mean_t(x) @ wm_w + wm_b)   (one block, 512 threads)
// ---------------------------------------------------------------------------
__global__ __launch_bounds__(512) void k_alpha(const float* __restrict__ x,
                                               const float* __restrict__ wmw,
                                               const float* __restrict__ wmb,
                                               float* __restrict__ alpha) {
    __shared__ float mean[DM];
    __shared__ float logit[4];
    const int d = threadIdx.x;
    float s = 0.f;
    for (int t = 0; t < SEQ; ++t) s += x[t * DM + d];
    mean[d] = s * (1.0f / SEQ);
    __syncthreads();
    if (d < 4) {
        float l = wmb[d];
        for (int c = 0; c < DM; ++c) l += mean[c] * wmw[c * 4 + d];
        logit[d] = l;
    }
    __syncthreads();
    if (d == 0) {
        float mx = fmaxf(fmaxf(logit[0], logit[1]), fmaxf(logit[2], logit[3]));
        float e0 = __expf(logit[0] - mx), e1 = __expf(logit[1] - mx);
        float e2 = __expf(logit[2] - mx), e3 = __expf(logit[3] - mx);
        float inv = 1.0f / (e0 + e1 + e2 + e3);
        alpha[0] = e0 * inv; alpha[1] = e1 * inv;
        alpha[2] = e2 * inv; alpha[3] = e3 * inv;
    }
}

// ---------------------------------------------------------------------------
// K1: effective conv weights: 4 gated dilated K=3 convs collapse into 9
// shift-offset matrices W_eff[off][o][c] (bf16) + effective bias.
// ---------------------------------------------------------------------------
__global__ __launch_bounds__(256) void k_weff(const float* __restrict__ convw,
                                              const float* __restrict__ convb,
                                              const float* __restrict__ alpha,
                                              u16* __restrict__ weff,
                                              float* __restrict__ beff) {
    const int OFFS[9] = {-8, -4, -2, -1, 0, 1, 2, 4, 8};
    const int DIL[4]  = {1, 2, 4, 8};
    const int b  = blockIdx.x;
    const int oi = b >> 9;
    const int o  = b & 511;
    const float al[4] = {alpha[0], alpha[1], alpha[2], alpha[3]};
    if (oi == 0 && threadIdx.x == 0) {
        beff[o] = al[0] * convb[o] + al[1] * convb[DM + o] +
                  al[2] * convb[2 * DM + o] + al[3] * convb[3 * DM + o];
    }
    const int off = OFFS[oi];
    for (int c = threadIdx.x; c < DM; c += 256) {
        float s = 0.f;
        #pragma unroll
        for (int i = 0; i < 4; ++i)
            #pragma unroll
            for (int k = 0; k < 3; ++k)
                if ((k - 1) * DIL[i] == off)
                    s += al[i] * convw[(((size_t)i * DM + o) * DM + c) * 3 + k];
        weff[((size_t)oi * DM + o) * DM + c] = f2bf(s);
    }
}

// x (f32 [1024,512]) -> zero-padded bf16 [1040,512]; pad rows 0..7, 1032..1039
__global__ __launch_bounds__(256) void k_cvt_pad(const float* __restrict__ src,
                                                 u16* __restrict__ dst) {
    const int i = blockIdx.x * 256 + threadIdx.x;     // over SEQP*DM
    const int row = i >> 9;
    u16 v = 0;
    if (row >= PAD && row < SEQ + PAD)
        v = f2bf(src[i - PAD * DM]);
    dst[i] = v;
}

// LDS-tiled transpose+convert: dst[n][k] = bf16(src[k][n]); src is K x N.
__global__ __launch_bounds__(256) void k_tcvt(const float* __restrict__ src,
                                              u16* __restrict__ dst, int K, int N) {
    __shared__ float tile[32][33];
    const int tk = blockIdx.x % (K >> 5);
    const int tn = blockIdx.x / (K >> 5);
    const int j  = threadIdx.x & 31;
    const int i0 = threadIdx.x >> 5;   // 0..7
    const int k0 = tk * 32, n0 = tn * 32;
    #pragma unroll
    for (int i = i0; i < 32; i += 8)
        tile[i][j] = src[(size_t)(k0 + i) * N + n0 + j];
    __syncthreads();
    #pragma unroll
    for (int i = i0; i < 32; i += 8)
        dst[(size_t)(n0 + i) * K + k0 + j] = f2bf(tile[j][i]);
}

// ---------------------------------------------------------------------------
// K2: conv as 9 shifted GEMMs on the zero-padded input: branch-free rotated
// 2-stage pipeline, kept in loop form so the stages occupy distinct registers.
// Wave = 16x64 strip; 512 waves = 64 blk x 8 wv.
// ---------------------------------------------------------------------------
__global__ __launch_bounds__(256) void k_conv_gemm(const u16* __restrict__ xpad,
                                                   const u16* __restrict__ weff,
                                                   float* __restrict__ hacc) {
    const int wv   = threadIdx.x >> 5;
    const int wave = blockIdx.x * 8 + wv;
    const int lane = threadIdx.x & 31;
    const int half = lane >> 4;
    const int lm   = lane & 15;
    const int m0   = (wave >> 3) * 16;
    const int n0   = (wave & 7) * 64;
    const int OFFS[9] = {-8, -4, -2, -1, 0, 1, 2, 4, 8};
    __shared__ float smem[8 * 1024];

    v8f z8 = {0.f, 0.f, 0.f, 0.f, 0.f, 0.f, 0.f, 0.f};
    v8f acc[4];
    #pragma unroll
    for (int t = 0; t < 4; ++t) acc[t] = z8;

    #pragma unroll 1
    for (int oi = 0; oi < 9; ++oi) {
        const int row = m0 + lm + PAD + OFFS[oi];     // always in [0, SEQP)
        const u16* W = weff + (size_t)oi * DM * DM;
        Stage s0, s1;
        load_stage(s0, xpad, row, DM, W, n0 + lm, DM, 0,  half);
        load_stage(s1, xpad, row, DM, W, n0 + lm, DM, 32, half);
        #pragma unroll 1
        for (int kb = 64; kb < DM; kb += 64) {
            fire(acc, s0);
            load_stage(s0, xpad, row, DM, W, n0 + lm, DM, kb, half);
            fire(acc, s1);
            load_stage(s1, xpad, row, DM, W, n0 + lm, DM, kb + 32, half);
        }
        fire(acc, s0);
        fire(acc, s1);
    }
    // LDS repack -> coalesced b128 stores
    float* my = smem + wv * 1024;
    #pragma unroll
    for (int t = 0; t < 4; ++t)
        #pragma unroll
        for (int v = 0; v < 8; ++v)
            my[(half * 8 + v) * 64 + t * 16 + lm] = acc[t][v];
    __syncthreads();
    for (int c = lane; c < 256; c += 32) {
        v4f val = ((const v4f*)my)[c];
        const int row = m0 + (c >> 4);
        const int col = n0 + (c & 15) * 4;
        *(v4f*)(hacc + (size_t)row * DM + col) = val;
    }
}

// ---------------------------------------------------------------------------
// K3: row-wise  u = hacc + b_eff + x ; h1 = LN1(u) ; z = LN2(h1)
// (SSM scalar addend cancels under LN2 -> output collapses to one row-set)
// ---------------------------------------------------------------------------
__global__ __launch_bounds__(256) void k_ln(const float* __restrict__ hacc,
                                            const float* __restrict__ beff,
                                            const float* __restrict__ x,
                                            const float* __restrict__ g1,
                                            const float* __restrict__ b1,
                                            const float* __restrict__ g2,
                                            const float* __restrict__ b2,
                                            float* __restrict__ zf,
                                            u16* __restrict__ zbf) {
    const int t = blockIdx.x, tid = threadIdx.x;
    __shared__ float red[256];
    const int i0 = t * DM + tid, i1 = i0 + 256;

    float u0 = hacc[i0] + beff[tid] + x[i0];
    float u1 = hacc[i1] + beff[tid + 256] + x[i1];

    red[tid] = u0 + u1; __syncthreads();
    for (int s = 128; s > 0; s >>= 1) { if (tid < s) red[tid] += red[tid + s]; __syncthreads(); }
    float m = red[0] * (1.0f / DM); __syncthreads();
    float d0 = u0 - m, d1 = u1 - m;
    red[tid] = d0 * d0 + d1 * d1; __syncthreads();
    for (int s = 128; s > 0; s >>= 1) { if (tid < s) red[tid] += red[tid + s]; __syncthreads(); }
    float rstd = rsqrtf(red[0] * (1.0f / DM) + 1e-5f); __syncthreads();
    float h0 = d0 * rstd * g1[tid] + b1[tid];
    float h1v = d1 * rstd * g1[tid + 256] + b1[tid + 256];

    red[tid] = h0 + h1v; __syncthreads();
    for (int s = 128; s > 0; s >>= 1) { if (tid < s) red[tid] += red[tid + s]; __syncthreads(); }
    m = red[0] * (1.0f / DM); __syncthreads();
    d0 = h0 - m; d1 = h1v - m;
    red[tid] = d0 * d0 + d1 * d1; __syncthreads();
    for (int s = 128; s > 0; s >>= 1) { if (tid < s) red[tid] += red[tid + s]; __syncthreads(); }
    rstd = rsqrtf(red[0] * (1.0f / DM) + 1e-5f);
    float z0 = d0 * rstd * g2[tid] + b2[tid];
    float z1 = d1 * rstd * g2[tid + 256] + b2[tid + 256];

    zf[i0] = z0; zf[i1] = z1;
    zbf[i0] = f2bf(z0); zbf[i1] = f2bf(z1);
}

// ---------------------------------------------------------------------------
// K4: GEMM1  act = relu(z @ W1 + b1) -> bf16.  2048 waves = 256 blk x 8 wv.
// ---------------------------------------------------------------------------
__global__ __launch_bounds__(256) void k_gemm1(const u16* __restrict__ zbf,
                                               const u16* __restrict__ w1t,
                                               const float* __restrict__ b1,
                                               u16* __restrict__ act) {
    const int wv   = threadIdx.x >> 5;
    const int wave = blockIdx.x * 8 + wv;
    const int lane = threadIdx.x & 31;
    const int half = lane >> 4;
    const int lm   = lane & 15;
    const int m0   = (wave >> 5) * 16;
    const int n0   = (wave & 31) * 64;
    __shared__ u16 smem16[8 * 1024];

    v8f z8 = {0.f, 0.f, 0.f, 0.f, 0.f, 0.f, 0.f, 0.f};
    v8f acc[4];
    #pragma unroll
    for (int t = 0; t < 4; ++t) acc[t] = z8;

    Stage s0, s1;
    load_stage(s0, zbf, m0 + lm, DM, w1t, n0 + lm, DM, 0,  half);
    load_stage(s1, zbf, m0 + lm, DM, w1t, n0 + lm, DM, 32, half);
    #pragma unroll 1
    for (int kb = 64; kb < DM; kb += 64) {
        fire(acc, s0);
        load_stage(s0, zbf, m0 + lm, DM, w1t, n0 + lm, DM, kb, half);
        fire(acc, s1);
        load_stage(s1, zbf, m0 + lm, DM, w1t, n0 + lm, DM, kb + 32, half);
    }
    fire(acc, s0);
    fire(acc, s1);

    // bias + relu + bf16 convert, LDS repack -> b128 stores
    u16* my = smem16 + wv * 1024;
    #pragma unroll
    for (int t = 0; t < 4; ++t) {
        const int col = n0 + t * 16 + lm;
        const float bias = b1[col];
        #pragma unroll
        for (int v = 0; v < 8; ++v)
            my[(half * 8 + v) * 64 + t * 16 + lm] = f2bf(fmaxf(acc[t][v] + bias, 0.0f));
    }
    __syncthreads();
    for (int c = lane; c < 128; c += 32) {
        v4u val = ((const v4u*)my)[c];
        const int row = m0 + (c >> 3);
        const int col = n0 + (c & 7) * 8;
        *(v4u*)(act + (size_t)row * DFF + col) = val;
    }
}

// ---------------------------------------------------------------------------
// K5: GEMM2  out_row = act @ W2 + b2 + z ; broadcast-store 64 batch copies
// via LDS repack + b128 stores.  512 waves = 64 blk x 8 wv, K=2048.
// ---------------------------------------------------------------------------
__global__ __launch_bounds__(256) void k_gemm2(const u16* __restrict__ act,
                                               const u16* __restrict__ w2t,
                                               const float* __restrict__ b2,
                                               const float* __restrict__ zf,
                                               float* __restrict__ out) {
    const int wv   = threadIdx.x >> 5;
    const int wave = blockIdx.x * 8 + wv;
    const int lane = threadIdx.x & 31;
    const int half = lane >> 4;
    const int lm   = lane & 15;
    const int m0   = (wave >> 3) * 16;
    const int n0   = (wave & 7) * 64;
    __shared__ float smem[8 * 1024];

    v8f z8 = {0.f, 0.f, 0.f, 0.f, 0.f, 0.f, 0.f, 0.f};
    v8f acc[4];
    #pragma unroll
    for (int t = 0; t < 4; ++t) acc[t] = z8;

    Stage s0, s1;
    load_stage(s0, act, m0 + lm, DFF, w2t, n0 + lm, DFF, 0,  half);
    load_stage(s1, act, m0 + lm, DFF, w2t, n0 + lm, DFF, 32, half);
    #pragma unroll 1
    for (int kb = 64; kb < DFF; kb += 64) {
        fire(acc, s0);
        load_stage(s0, act, m0 + lm, DFF, w2t, n0 + lm, DFF, kb, half);
        fire(acc, s1);
        load_stage(s1, act, m0 + lm, DFF, w2t, n0 + lm, DFF, kb + 32, half);
    }
    fire(acc, s0);
    fire(acc, s1);

    // bias + residual, LDS repack
    float* my = smem + wv * 1024;
    #pragma unroll
    for (int t = 0; t < 4; ++t) {
        const int col = n0 + t * 16 + lm;
        const float bias = b2[col];
        #pragma unroll
        for (int v = 0; v < 8; ++v) {
            const int row = m0 + half * 8 + v;
            my[(half * 8 + v) * 64 + t * 16 + lm] =
                acc[t][v] + bias + zf[(size_t)row * DM + col];
        }
    }
    __syncthreads();
    // broadcast 64 batch copies with coalesced b128 stores
    const size_t bstride = (size_t)SEQ * DM;
    for (int c = lane; c < 256; c += 32) {
        v4f val = ((const v4f*)my)[c];
        const int row = m0 + (c >> 4);
        const int col = n0 + (c & 15) * 4;
        const size_t base = (size_t)row * DM + col;
        #pragma unroll 8
        for (int i = 0; i < NBATCH; ++i)
            *(v4f*)(out + base + (size_t)i * bstride) = val;
    }
}

// ---------------------------------------------------------------------------
extern "C" void kernel_launch(void* const* d_in, const int* in_sizes, int n_in,
                              void* d_out, int out_size, void* d_ws, size_t ws_size,
                              hipStream_t stream) {
    const float* x     = (const float*)d_in[0];   // [1,1024,512]
    const float* convw = (const float*)d_in[1];   // [4,512,512,3]
    const float* convb = (const float*)d_in[2];   // [4,512]
    const float* wmw   = (const float*)d_in[3];   // [512,4]
    const float* wmb   = (const float*)d_in[4];   // [4]
    // d_in[5..11]: delta/B/C/A_log — SSM output is a scalar per (i,t) broadcast
    // over the feature dim; LayerNorm cancels it, so it cannot affect d_out.
    const float* ln1g  = (const float*)d_in[12];
    const float* ln1b  = (const float*)d_in[13];
    const float* ln2g  = (const float*)d_in[14];
    const float* ln2b  = (const float*)d_in[15];
    const float* w1    = (const float*)d_in[16];  // [512,2048]
    const float* fb1   = (const float*)d_in[17];  // [2048]
    const float* w2    = (const float*)d_in[18];  // [2048,512]
    const float* fb2   = (const float*)d_in[19];  // [512]
    float* out = (float*)d_out;

    // workspace carve-up (all chunks 16B aligned)
    float* alpha = (float*)d_ws;                  // 4
    float* beff  = alpha + 4;                     // 512
    float* hacc  = beff + 512;                    // 1024*512
    float* zf    = hacc + SEQ * DM;               // 1024*512
    u16*   xpad  = (u16*)(zf + SEQ * DM);         // 1040*512 (zero-padded)
    u16*   zbf   = xpad + SEQP * DM;              // 1024*512
    u16*   weff  = zbf + SEQ * DM;                // 9*512*512
    u16*   w1t   = weff + 9 * DM * DM;            // 2048*512
    u16*   w2t   = w1t + DFF * DM;                // 512*2048
    u16*   act   = w2t + DM * DFF;                // 1024*2048

    k_alpha<<<1, 512, 0, stream>>>(x, wmw, wmb, alpha);
    k_weff<<<9 * 512, 256, 0, stream>>>(convw, convb, alpha, weff, beff);
    k_cvt_pad<<<(SEQP * DM) / 256, 256, 0, stream>>>(x, xpad);
    // W1^T : [2048][512]   (K=512, N=2048) -> 1024 tiles
    k_tcvt<<<(DM / 32) * (DFF / 32), 256, 0, stream>>>(w1, w1t, DM, DFF);
    // W2^T : [512][2048]   (K=2048, N=512) -> 1024 tiles
    k_tcvt<<<(DFF / 32) * (DM / 32), 256, 0, stream>>>(w2, w2t, DFF, DM);

    k_conv_gemm<<<64, 256, 0, stream>>>(xpad, weff, hacc);
    k_ln<<<SEQ, 256, 0, stream>>>(hacc, beff, x, ln1g, ln1b, ln2g, ln2b, zf, zbf);
    k_gemm1<<<256, 256, 0, stream>>>(zbf, w1t, fb1, act);
    k_gemm2<<<64, 256, 0, stream>>>(act, w2t, fb2, zf, out);
}
